// Head_35485019800116
// MI455X (gfx1250) — matrix operand order repaired and checked by reading, not compiled, not run on hardware
//
#include <hip/hip_runtime.h>

// ---------------------------------------------------------------------------
// Fused QKV projection + causal flash attention for MI455X (gfx1250, wave32).
//   Pass 0: f32 -> bf16 conversion of x and W (bandwidth-bound, ~4us @23.3TB/s)
//   Pass 1: qkv GEMM, A tile staged in LDS via TENSOR_LOAD_TO_LDS (TDM),
//           all GEMM math in V_WMMA_F32_16X16X32_BF16 (f32 accumulate)
//   Pass 2: causal flash attention, WMMA for S=QK^T and O+=PV
// ---------------------------------------------------------------------------

typedef __attribute__((ext_vector_type(16))) __bf16 v16bf;
typedef __attribute__((ext_vector_type(8)))  __bf16 v8bf;
typedef __attribute__((ext_vector_type(8)))  float  v8f;
typedef __attribute__((ext_vector_type(4)))  unsigned int v4u32;
typedef __attribute__((ext_vector_type(8)))  int v8i32;
typedef __attribute__((ext_vector_type(4)))  int v4i32;

constexpr int kB   = 8;
constexpr int kT   = 2048;
constexpr int kD   = 1024;
constexpr int kHD  = 128;
constexpr int kNQK = 384;                       // 3 * kHD rows of W_qkv

static __device__ __forceinline__ v8f wmma_bf16(v16bf a, v16bf b, v8f c) {
  return __builtin_amdgcn_wmma_f32_16x16x32_bf16(false, a, false, b,
                                                 (short)0, c, false, false);
}

// ---------------------------------------------------------------------------
// TDM: DMA a [rows x row_elems] bf16 tile (contiguous rows, stride=row_elems)
// from global memory into LDS.  D# bit layout per CDNA5 ISA ch.8.
// This toolchain exposes the 6-arg builtin (clang-23 / therock-10.0 form).
// ---------------------------------------------------------------------------
static __device__ __forceinline__ void tdm_load_tile_bf16(
    unsigned int lds_byte_addr, const void* gaddr,
    unsigned int row_elems, unsigned int rows)
{
  const unsigned long long ga = (unsigned long long)(uintptr_t)gaddr;
  v4u32 g0;
  g0[0] = 1u;                                        // count=1 (valid), user D#
  g0[1] = lds_byte_addr;                             // lds_addr [63:32]
  g0[2] = (unsigned int)(ga & 0xFFFFFFFFu);          // global_addr [95:64]
  g0[3] = (unsigned int)((ga >> 32) & 0x01FFFFFFu)   // global_addr[56:32]
        | (2u << 30);                                // type=2 ("image")
  v8i32 g1;
  g1[0] = (int)(1u << 16);                           // data_size=1 (2B), mask=0
  g1[1] = (int)((row_elems & 0xFFFFu) << 16);        // tensor_dim0 lo @63:48
  g1[2] = (int)(((row_elems >> 16) & 0xFFFFu)        // tensor_dim0 hi @79:64
        | ((rows & 0xFFFFu) << 16));                 // tensor_dim1 lo @95:80
  g1[3] = (int)(((rows >> 16) & 0xFFFFu)             // tensor_dim1 hi @111:96
        | ((row_elems & 0xFFFFu) << 16));            // tile_dim0 @127:112
  g1[4] = (int)(rows & 0xFFFFu);                     // tile_dim1 @143:128
  g1[5] = (int)row_elems;                            // tensor_dim0_stride lo32
  g1[6] = 0;                                         // stride hi, dim1_stride lo
  g1[7] = 0;
  v4i32 gz4 = {0, 0, 0, 0};
  v8i32 gz8 = {0, 0, 0, 0, 0, 0, 0, 0};
  __builtin_amdgcn_tensor_load_to_lds(g0, g1, gz4, gz4, gz8, 0);
}

// ---------------------------------------------------------------------------
// Pass 0: convert x (B*T*D) and W (384*D) from f32 to bf16, 8 elems/thread.
// ---------------------------------------------------------------------------
__global__ __launch_bounds__(256) void cvt_bf16_kernel(
    const float* __restrict__ x, const float* __restrict__ W,
    __bf16* __restrict__ xb, __bf16* __restrict__ wb)
{
  const size_t xN = (size_t)kB * kT * kD;
  const size_t wN = (size_t)kNQK * kD;
  size_t i = ((size_t)blockIdx.x * blockDim.x + threadIdx.x) * 8;
  const float* src;
  __bf16* dst;
  if (i < xN) { src = x + i; dst = xb + i; }
  else {
    size_t j = i - xN;
    if (j >= wN) return;
    src = W + j; dst = wb + j;
  }
  float4 f0 = *(const float4*)(src);
  float4 f1 = *(const float4*)(src + 4);
  v8bf o;
  o[0]=(__bf16)f0.x; o[1]=(__bf16)f0.y; o[2]=(__bf16)f0.z; o[3]=(__bf16)f0.w;
  o[4]=(__bf16)f1.x; o[5]=(__bf16)f1.y; o[6]=(__bf16)f1.z; o[7]=(__bf16)f1.w;
  *(v8bf*)dst = o;
}

// ---------------------------------------------------------------------------
// Pass 1: qkv = x @ W^T.  Block = 8 waves sharing one 16-row A tile (32 KB in
// LDS via TDM); each wave owns one 16-wide slice of the 128-col N panel.
// Writes q (pre-scaled by 1/sqrt(128)) / k as bf16 [b][t][128],
// v transposed as bf16 [b][128][t].
// ---------------------------------------------------------------------------
__global__ __launch_bounds__(256) void qkv_gemm_kernel(
    const __bf16* __restrict__ xb, const __bf16* __restrict__ wb,
    __bf16* __restrict__ qo, __bf16* __restrict__ ko, __bf16* __restrict__ vTo)
{
  __shared__ __bf16 Atile[16 * kD];               // 32 KB
  const int lane  = threadIdx.x & 31;
  const int wave  = threadIdx.x >> 5;
  const int mtile = blockIdx.x / 3;               // 1024 row tiles
  const int ntile = (blockIdx.x % 3) * 8 + wave;  // 24 col tiles
  const int half  = lane >> 4;
  const int ln    = lane & 15;

  // DMA the A tile (16 x 1024 bf16, contiguous) into LDS with the TDM.
  if (wave == 0) {
    tdm_load_tile_bf16((unsigned int)(uintptr_t)&Atile[0],
                       xb + (size_t)mtile * 16 * kD, kD, 16);
    __builtin_amdgcn_s_wait_tensorcnt(0);
  }
  __syncthreads();

  const __bf16* arow = Atile + (size_t)ln * kD;
  const __bf16* brow = wb + (size_t)(ntile * 16 + ln) * kD;

  v8f acc = {};
  for (int d0 = 0; d0 < kD; d0 += 32) {
    // A fragment (16x32): lanes 0-15 K=0..15, lanes 16-31 K=16..31
    v8bf alo = *(const v8bf*)(arow + d0 + half * 16);
    v8bf ahi = *(const v8bf*)(arow + d0 + half * 16 + 8);
    v16bf av;
    #pragma unroll
    for (int i = 0; i < 8; ++i) { av[i] = alo[i]; av[i + 8] = ahi[i]; }
    // B fragment (32x16): lanes 0-15 K=0..15, lanes 16-31 K=16..31
    v16bf bv = *(const v16bf*)(brow + d0 + half * 16);
    acc = wmma_bf16(av, bv, acc);
  }

  // Epilogue: selector is uniform per wave -> scalarize for s_cbranch.
  const int   nsel  = __builtin_amdgcn_readfirstlane(ntile);
  const float scale = 0.08838834764831845f;       // 1/sqrt(128) folded into q
  if (nsel < 8) {                                 // ---- Q ----
    const int h = nsel * 16 + ln;
    #pragma unroll
    for (int r = 0; r < 8; ++r) {
      const int token = mtile * 16 + r + half * 8;
      qo[(size_t)token * kHD + h] = (__bf16)(acc[r] * scale);
    }
  } else if (nsel < 16) {                         // ---- K ----
    const int h = (nsel - 8) * 16 + ln;
    #pragma unroll
    for (int r = 0; r < 8; ++r) {
      const int token = mtile * 16 + r + half * 8;
      ko[(size_t)token * kHD + h] = (__bf16)acc[r];
    }
  } else {                                        // ---- V^T ----
    const int h = (nsel - 16) * 16 + ln;
    #pragma unroll
    for (int r = 0; r < 8; ++r) {
      const int token = mtile * 16 + r + half * 8;
      const int b = token >> 11;
      const int t = token & (kT - 1);
      vTo[((size_t)(b * kHD + h)) * kT + t] = (__bf16)acc[r];
    }
  }
}

// ---------------------------------------------------------------------------
// Pass 2: causal flash attention.  One wave owns a 16-query block.
// Per 32-key tile: S = Q K^T (8 wmma), mask + online softmax (f32, shfl_xor
// within 16-lane halves), P relayout C->A through LDS, O += P V (8 wmma).
// ---------------------------------------------------------------------------
__global__ __launch_bounds__(128) void flash_attn_kernel(
    const __bf16* __restrict__ qg, const __bf16* __restrict__ kg,
    const __bf16* __restrict__ vTg, float* __restrict__ out)
{
  __shared__ __bf16 lds_p[4 * 16 * 32];           // per-wave P staging

  const int lane = threadIdx.x & 31;
  const int wave = threadIdx.x >> 5;
  const int gw   = blockIdx.x * 4 + wave;         // 8 * 128 q-tiles
  const int b    = gw >> 7;
  const int q0   = (gw & 127) * 16;
  const int half = lane >> 4;
  const int ln   = lane & 15;

  const __bf16* qb = qg  + (size_t)b * kT * kHD;
  const __bf16* kb = kg  + (size_t)b * kT * kHD;
  const __bf16* vb = vTg + (size_t)b * kHD * kT;

  // Q fragments: 4 slices of K=32 over head dim 128 (A layout)
  v16bf qa[4];
  {
    const __bf16* qrow = qb + (size_t)(q0 + ln) * kHD;
    #pragma unroll
    for (int dt = 0; dt < 4; ++dt) {
      const int aoff = dt * 32 + half * 16;
      v8bf lo = *(const v8bf*)(qrow + aoff);
      v8bf hi = *(const v8bf*)(qrow + aoff + 8);
      v16bf f;
      #pragma unroll
      for (int i = 0; i < 8; ++i) { f[i] = lo[i]; f[i + 8] = hi[i]; }
      qa[dt] = f;
    }
  }

  v8f o[8] = {};
  float mrow[8], lrow[8];
  #pragma unroll
  for (int r = 0; r < 8; ++r) { mrow[r] = -__builtin_inff(); lrow[r] = 0.f; }

  __bf16* pbuf = lds_p + wave * (16 * 32);

  for (int k0 = 0; k0 <= q0 + 15; k0 += 32) {
    // Prefetch next key tile while this one computes.
    if (k0 + 63 < kT) {
      __builtin_prefetch(kb + (size_t)(k0 + 32 + ln) * kHD, 0, 3);
      __builtin_prefetch(vb + (size_t)ln * kT + k0 + 32, 0, 3);
    }

    // ---- S = Q K^T for two 16-key column tiles -------------------------
    v8f s0 = {}, s1 = {};
    {
      const __bf16* krow0 = kb + (size_t)(k0 + ln) * kHD;
      const __bf16* krow1 = krow0 + 16 * kHD;
      #pragma unroll
      for (int dt = 0; dt < 4; ++dt) {
        const int boff = dt * 32 + half * 16;
        v16bf b0 = *(const v16bf*)(krow0 + boff);
        v16bf b1 = *(const v16bf*)(krow1 + boff);
        s0 = wmma_bf16(qa[dt], b0, s0);
        s1 = wmma_bf16(qa[dt], b1, s1);
      }
    }

    // ---- causal mask + online softmax (rows live in 16-lane halves) ----
    const int col0 = k0 + ln;
    const int col1 = col0 + 16;
    float p0[8], p1[8], alpha[8];
    #pragma unroll
    for (int r = 0; r < 8; ++r) {
      const int qi = q0 + r + half * 8;
      const float v0 = (col0 <= qi) ? s0[r] : -__builtin_inff();
      const float v1 = (col1 <= qi) ? s1[r] : -__builtin_inff();
      float rm = fmaxf(v0, v1);
      rm = fmaxf(rm, __shfl_xor(rm, 1, 32));
      rm = fmaxf(rm, __shfl_xor(rm, 2, 32));
      rm = fmaxf(rm, __shfl_xor(rm, 4, 32));
      rm = fmaxf(rm, __shfl_xor(rm, 8, 32));
      const float mnew = fmaxf(mrow[r], rm);
      const float a  = __expf(mrow[r] - mnew);
      const float e0 = __expf(v0 - mnew);
      const float e1 = __expf(v1 - mnew);
      float rs = e0 + e1;
      rs += __shfl_xor(rs, 1, 32);
      rs += __shfl_xor(rs, 2, 32);
      rs += __shfl_xor(rs, 4, 32);
      rs += __shfl_xor(rs, 8, 32);
      lrow[r] = lrow[r] * a + rs;
      mrow[r] = mnew;
      alpha[r] = a;
      p0[r] = e0; p1[r] = e1;
    }
    #pragma unroll
    for (int ht = 0; ht < 8; ++ht)
      #pragma unroll
      for (int r = 0; r < 8; ++r)
        o[ht][r] *= alpha[r];

    // ---- P: C-layout -> A-layout via per-wave LDS tile -----------------
    #pragma unroll
    for (int r = 0; r < 8; ++r) {
      const int row = r + half * 8;
      pbuf[row * 32 + ln]      = (__bf16)p0[r];
      pbuf[row * 32 + ln + 16] = (__bf16)p1[r];
    }
    v16bf pa;
    {
      const __bf16* prow = pbuf + ln * 32 + half * 16;
      v8bf lo = *(const v8bf*)(prow);
      v8bf hi = *(const v8bf*)(prow + 8);
      #pragma unroll
      for (int i = 0; i < 8; ++i) { pa[i] = lo[i]; pa[i + 8] = hi[i]; }
    }

    // ---- O += P @ V  (V stored transposed: contiguous B fragments) ----
    #pragma unroll
    for (int ht = 0; ht < 8; ++ht) {
      const __bf16* vrow = vb + (size_t)(ht * 16 + ln) * kT + k0 + half * 16;
      v16bf bv = *(const v16bf*)(vrow);
      o[ht] = wmma_bf16(pa, bv, o[ht]);
    }
  }

  // ---- epilogue: O /= l ------------------------------------------------
  float* ob = out + (size_t)(b * kT + q0) * kHD;
  #pragma unroll
  for (int r = 0; r < 8; ++r) {
    const float inv = 1.0f / lrow[r];
    const int row = r + half * 8;
    #pragma unroll
    for (int ht = 0; ht < 8; ++ht)
      ob[(size_t)row * kHD + ht * 16 + ln] = o[ht][r] * inv;
  }
}

// ---------------------------------------------------------------------------
extern "C" void kernel_launch(void* const* d_in, const int* in_sizes, int n_in,
                              void* d_out, int out_size, void* d_ws, size_t ws_size,
                              hipStream_t stream) {
  const float* x = (const float*)d_in[0];   // (8, 2048, 1024) f32
  const float* W = (const float*)d_in[1];   // (384, 1024) f32
  float* out = (float*)d_out;               // (8, 2048, 128) f32

  const size_t xN = (size_t)kB * kT * kD;       // 16,777,216
  const size_t wN = (size_t)kNQK * kD;          //    393,216
  const size_t qN = (size_t)kB * kT * kHD;      //  2,097,152

  __bf16* xb  = (__bf16*)d_ws;                  // 32 MiB
  __bf16* wbp = xb  + xN;                       // 768 KiB
  __bf16* wsq = wbp + wN;                       // 4 MiB
  __bf16* wsk = wsq + qN;                       // 4 MiB
  __bf16* wsv = wsk + qN;                       // 4 MiB (transposed V)

  // Pass 0: (xN + wN) / 8 / 256 = 8384 blocks exactly
  cvt_bf16_kernel<<<8384, 256, 0, stream>>>(x, W, xb, wbp);
  // Pass 1: 1024 M-tiles * 3 N-panels, 8 waves (16-col slices) per block
  qkv_gemm_kernel<<<3072, 256, 0, stream>>>(xb, wbp, wsq, wsk, wsv);
  // Pass 2: 8 batches * 128 q-tiles, 4 waves per 128-thread block
  flash_attn_kernel<<<256, 128, 0, stream>>>(wsq, wsk, wsv, out);
}
